// MHAPyTorchScaledDotProduct_3143916060978
// MI455X (gfx1250) — matrix operand order repaired and checked by reading, not compiled, not run on hardware
//
#include <hip/hip_runtime.h>
#include <hip/hip_bf16.h>
#include <stdint.h>

typedef __attribute__((ext_vector_type(16))) __bf16 v16bf;
typedef __attribute__((ext_vector_type(8)))  float  v8f;

#define D_IN   1024
#define D_OUTD 1024
#define NHEAD  16
#define HDIM   64
#define BATCH  2
#define SEQ    2048
#define MTOT   (BATCH * SEQ)

static __device__ __forceinline__ uint16_t f2bf(float f) {
  uint32_t u = __float_as_uint(f);
  u += 0x7FFFu + ((u >> 16) & 1u);
  return (uint16_t)(u >> 16);
}

static __device__ __forceinline__ v8f zero8() {
  v8f z;
#pragma unroll
  for (int i = 0; i < 8; ++i) z[i] = 0.0f;
  return z;
}

static __device__ __forceinline__ v8f wmma_bf16(v16bf a, v16bf b, v8f c) {
  return __builtin_amdgcn_wmma_f32_16x16x32_bf16(false, a, false, b, (short)0, c,
                                                 false, false);
}

// ---- CDNA5 async global->LDS copy (ASYNCcnt path), 16 bytes per lane ----
#if defined(__AMDGCN__)
typedef int v4i_ __attribute__((vector_size(16)));
typedef __attribute__((address_space(1))) v4i_ g_v4i;
typedef __attribute__((address_space(3))) v4i_ l_v4i;
#endif

static __device__ __forceinline__ void async_copy_b128(const void* gsrc, void* ldst) {
#if defined(__AMDGCN__)
#if __has_builtin(__builtin_amdgcn_global_load_async_to_lds_b128)
  __builtin_amdgcn_global_load_async_to_lds_b128(
      (g_v4i*)(uintptr_t)gsrc, (l_v4i*)(uintptr_t)ldst, 0, 0);
#else
  asm volatile("global_load_async_to_lds_b128 %0, %1, off"
               :: "v"((uint32_t)(uintptr_t)ldst), "v"((uint64_t)(uintptr_t)gsrc)
               : "memory");
#endif
#endif
}

static __device__ __forceinline__ void wait_async0() {
#if defined(__AMDGCN__)
#if __has_builtin(__builtin_amdgcn_s_wait_asynccnt)
  __builtin_amdgcn_s_wait_asynccnt(0);
#else
  asm volatile("s_wait_asynccnt 0x0" ::: "memory");
#endif
#endif
}

// A fragment 16x32 (MxK) from LDS tile stored [m][k] (stride in halves).
static __device__ __forceinline__ v16bf frag_a(const uint16_t* t, int stride, int lane) {
  const int row = lane & 15;
  const int kh  = (lane >> 4) << 4;
  union { uint32_t u[8]; v16bf v; } f;
  const uint16_t* rp = t + row * stride;
#pragma unroll
  for (int p = 0; p < 8; ++p) {
    const int k = 2 * p + kh;
    f.u[p] = *(const uint32_t*)(rp + k);
  }
  return f.v;
}

// B fragment 32x16 (KxN) from LDS tile stored [n][k] (B^T / weight layout).
static __device__ __forceinline__ v16bf frag_b_kmajor(const uint16_t* t, int stride, int lane) {
  const int n  = lane & 15;
  const int kg = (lane >> 4) << 4;
  union { uint32_t u[8]; v16bf v; } f;
  const uint16_t* rp = t + n * stride + kg;
#pragma unroll
  for (int p = 0; p < 8; ++p) f.u[p] = *(const uint32_t*)(rp + 2 * p);
  return f.v;
}

// B fragment 32x16 (KxN) from LDS tile stored [k][n] (V tile).
static __device__ __forceinline__ v16bf frag_b_nmajor(const uint16_t* t, int stride, int lane) {
  const int n  = lane & 15;
  const int kg = (lane >> 4) << 4;
  union { uint16_t h[16]; v16bf v; } f;
#pragma unroll
  for (int p = 0; p < 8; ++p) {
    const int k = kg + 2 * p;
    f.h[2 * p]     = t[k * stride + n];
    f.h[2 * p + 1] = t[(k + 1) * stride + n];
  }
  return f.v;
}

// ---------------------------------------------------------------- convert
__global__ void k_f32_to_bf16(const float* __restrict__ in, uint16_t* __restrict__ out, int n) {
  int i = blockIdx.x * blockDim.x + threadIdx.x;
  for (; i < n; i += gridDim.x * blockDim.x) out[i] = f2bf(in[i]);
}

// ---------------------------------------------------------------- GEMM
// C[M,N] = A[M,K] bf16 x W[N,K]^T bf16. Block 128x128, 8 waves, wave = 32x64.
// Double-buffered async global->LDS staging.
#define BM 128
#define BN 128
#define BKK 32
#define AST 40
#define BST 40

static __device__ __forceinline__ void stage_gemm(const uint16_t* __restrict__ A,
                                                  const uint16_t* __restrict__ W,
                                                  uint16_t* As, uint16_t* Bs,
                                                  int m0, int n0, int K, int k0, int tid) {
  const int r = tid >> 2;        // 0..63
  const int c = (tid & 3) * 8;   // 0,8,16,24
#pragma unroll
  for (int it = 0; it < 2; ++it) {
    const int row = r + 64 * it;
    async_copy_b128(A + (size_t)(m0 + row) * K + k0 + c, &As[row * AST + c]);
    async_copy_b128(W + (size_t)(n0 + row) * K + k0 + c, &Bs[row * BST + c]);
  }
}

__launch_bounds__(256, 2)
__global__ void k_gemm(const uint16_t* __restrict__ A, const uint16_t* __restrict__ W,
                       uint16_t* __restrict__ Cb, float* __restrict__ Cf,
                       const float* __restrict__ bias, int M, int N, int K) {
  __shared__ __align__(16) uint16_t As[2][BM * AST];
  __shared__ __align__(16) uint16_t Bs[2][BN * BST];
  const int tid  = threadIdx.x;
  const int lane = tid & 31;
  const int wave = tid >> 5;
  const int wr   = wave & 3;   // 4 wave-rows of 32
  const int wc   = wave >> 2;  // 2 wave-cols of 64
  const int m0   = blockIdx.x * BM;
  const int n0   = blockIdx.y * BN;

  v8f acc[2][4];
#pragma unroll
  for (int i = 0; i < 2; ++i)
#pragma unroll
    for (int j = 0; j < 4; ++j) acc[i][j] = zero8();

  stage_gemm(A, W, As[0], Bs[0], m0, n0, K, 0, tid);
  wait_async0();
  __syncthreads();

  int buf = 0;
  for (int k0 = 0; k0 < K; k0 += BKK) {
    if (k0 + BKK < K)
      stage_gemm(A, W, As[buf ^ 1], Bs[buf ^ 1], m0, n0, K, k0 + BKK, tid);

    v16bf af0 = frag_a(&As[buf][(wr * 32) * AST], AST, lane);
    v16bf af1 = frag_a(&As[buf][(wr * 32 + 16) * AST], AST, lane);
#pragma unroll
    for (int tn = 0; tn < 4; ++tn) {
      v16bf bfr = frag_b_kmajor(&Bs[buf][(wc * 64 + tn * 16) * BST], BST, lane);
      acc[0][tn] = wmma_bf16(af0, bfr, acc[0][tn]);
      acc[1][tn] = wmma_bf16(af1, bfr, acc[1][tn]);
    }

    wait_async0();
    __syncthreads();
    buf ^= 1;
  }

  const int hi = lane >> 4;
  const int nl = lane & 15;
#pragma unroll
  for (int tm = 0; tm < 2; ++tm)
#pragma unroll
    for (int tn = 0; tn < 4; ++tn)
#pragma unroll
      for (int v = 0; v < 8; ++v) {
        const int m = m0 + wr * 32 + tm * 16 + v + 8 * hi;
        const int n = n0 + wc * 64 + tn * 16 + nl;
        const float val = acc[tm][tn][v];
        if (Cf) Cf[(size_t)m * N + n] = val + (bias ? bias[n] : 0.0f);
        else    Cb[(size_t)m * N + n] = f2bf(val);
      }
}

// ---------------------------------------------------------------- flash attention
// qkv: [B,S,3*D_OUT] bf16, e = which*1024 + h*64 + d. Block = one (b,h) x 64 queries.
#define FBQ 64
#define FBK 64
#define FST 72

static __device__ __forceinline__ void stage_kv(const uint16_t* kbase, const uint16_t* vbase,
                                                uint16_t* Ks, uint16_t* Vs,
                                                int kb, size_t rs, int tid) {
  for (int i = tid; i < FBK * 8; i += 128) {
    const int row = i >> 3, cc = (i & 7) * 8;
    async_copy_b128(kbase + (size_t)(kb * FBK + row) * rs + cc, &Ks[row * FST + cc]);
    async_copy_b128(vbase + (size_t)(kb * FBK + row) * rs + cc, &Vs[row * FST + cc]);
  }
}

__launch_bounds__(128, 2)
__global__ void k_flash(const uint16_t* __restrict__ qkv, uint16_t* __restrict__ ctx) {
  __shared__ __align__(16) uint16_t Qs[FBQ * FST];
  __shared__ __align__(16) uint16_t Ks[2][FBK * FST];
  __shared__ __align__(16) uint16_t Vs[2][FBK * FST];
  __shared__ __align__(16) uint16_t Ps[4][16 * FST];

  const int tid  = threadIdx.x;
  const int lane = tid & 31;
  const int wave = tid >> 5;
  const int b    = blockIdx.x / NHEAD;
  const int h    = blockIdx.x % NHEAD;
  const int qb   = blockIdx.y * FBQ;
  const float scale = 0.125f;  // 1/sqrt(64)
  const size_t rs = 3 * D_OUTD;

  const uint16_t* qbase = qkv + (size_t)b * SEQ * rs + h * HDIM;
  const uint16_t* kbase = qbase + D_OUTD;
  const uint16_t* vbase = qbase + 2 * D_OUTD;

  for (int i = tid; i < FBQ * 8; i += 128) {
    const int row = i >> 3, cc = (i & 7) * 8;
    async_copy_b128(qbase + (size_t)(qb + row) * rs + cc, &Qs[row * FST + cc]);
  }
  stage_kv(kbase, vbase, Ks[0], Vs[0], 0, rs, tid);
  wait_async0();
  __syncthreads();

  v8f o[4];
  float mrow[8], lrow[8];
#pragma unroll
  for (int t = 0; t < 4; ++t) o[t] = zero8();
#pragma unroll
  for (int v = 0; v < 8; ++v) { mrow[v] = -3.0e38f; lrow[v] = 0.0f; }

  const int hi = lane >> 4;
  const int nl = lane & 15;
  const int kblocks = blockIdx.y + 1;  // causal
  int buf = 0;

  for (int kb = 0; kb < kblocks; ++kb) {
    if (kb + 1 < kblocks)
      stage_kv(kbase, vbase, Ks[buf ^ 1], Vs[buf ^ 1], kb + 1, rs, tid);

    // S = Q K^T (16 rows x 64 keys per wave)
    v8f sc[4];
#pragma unroll
    for (int t = 0; t < 4; ++t) sc[t] = zero8();
#pragma unroll
    for (int ks = 0; ks < 2; ++ks) {
      v16bf aq = frag_a(&Qs[(wave * 16) * FST + ks * 32], FST, lane);
#pragma unroll
      for (int tn = 0; tn < 4; ++tn) {
        v16bf bk = frag_b_kmajor(&Ks[buf][(tn * 16) * FST + ks * 32], FST, lane);
        sc[tn] = wmma_bf16(aq, bk, sc[tn]);
      }
    }

    const bool diag = (kb == (int)blockIdx.y);
#pragma unroll
    for (int tn = 0; tn < 4; ++tn)
#pragma unroll
      for (int v = 0; v < 8; ++v) {
        float s = sc[tn][v] * scale;
        if (diag) {
          const int m = qb + wave * 16 + v + 8 * hi;
          const int n = kb * FBK + tn * 16 + nl;
          if (n > m) s = -3.0e38f;
        }
        sc[tn][v] = s;
      }

    // online softmax; row r = v + 8*hi lives across each 16-lane half
#pragma unroll
    for (int v = 0; v < 8; ++v) {
      float mx = fmaxf(fmaxf(sc[0][v], sc[1][v]), fmaxf(sc[2][v], sc[3][v]));
#pragma unroll
      for (int off = 1; off < 16; off <<= 1) mx = fmaxf(mx, __shfl_xor(mx, off, 32));
      const float mnew  = fmaxf(mrow[v], mx);
      const float alpha = __expf(mrow[v] - mnew);
      float rsum = 0.0f;
#pragma unroll
      for (int tn = 0; tn < 4; ++tn) {
        const float p = __expf(sc[tn][v] - mnew);
        sc[tn][v] = p;
        rsum += p;
      }
#pragma unroll
      for (int off = 1; off < 16; off <<= 1) rsum += __shfl_xor(rsum, off, 32);
      lrow[v] = lrow[v] * alpha + rsum;
      mrow[v] = mnew;
#pragma unroll
      for (int tn = 0; tn < 4; ++tn) o[tn][v] *= alpha;
    }

    // P (C-layout) -> LDS -> A-layout for the PV matmul
#pragma unroll
    for (int tn = 0; tn < 4; ++tn)
#pragma unroll
      for (int v = 0; v < 8; ++v)
        Ps[wave][(v + 8 * hi) * FST + tn * 16 + nl] = f2bf(sc[tn][v]);

#pragma unroll
    for (int ks = 0; ks < 2; ++ks) {
      v16bf ap = frag_a(&Ps[wave][ks * 32], FST, lane);
#pragma unroll
      for (int tn = 0; tn < 4; ++tn) {
        v16bf bv = frag_b_nmajor(&Vs[buf][(ks * 32) * FST + tn * 16], FST, lane);
        o[tn] = wmma_bf16(ap, bv, o[tn]);
      }
    }

    wait_async0();
    __syncthreads();
    buf ^= 1;
  }

  // write ctx[b, s, h*64+d] as bf16
#pragma unroll
  for (int tn = 0; tn < 4; ++tn)
#pragma unroll
    for (int v = 0; v < 8; ++v) {
      const int m = qb + wave * 16 + v + 8 * hi;
      const int d = tn * 16 + nl;
      const float val = o[tn][v] / lrow[v];
      ctx[((size_t)(b * SEQ + m)) * D_OUTD + h * HDIM + d] = f2bf(val);
    }
}

// ---------------------------------------------------------------- launch
extern "C" void kernel_launch(void* const* d_in, const int* in_sizes, int n_in,
                              void* d_out, int out_size, void* d_ws, size_t ws_size,
                              hipStream_t stream) {
  const float* x      = (const float*)d_in[0];
  const float* w_qkv  = (const float*)d_in[1];
  const float* w_proj = (const float*)d_in[2];
  const float* b_proj = (const float*)d_in[3];
  float* out = (float*)d_out;

  char* ws = (char*)d_ws;
  size_t off = 0;
  auto carve = [&](size_t bytes) -> char* {
    char* p = ws + off;
    off = (off + bytes + 255) & ~(size_t)255;
    return p;
  };
  uint16_t* xb   = (uint16_t*)carve((size_t)MTOT * D_IN * 2);
  uint16_t* wqb  = (uint16_t*)carve((size_t)3 * D_OUTD * D_IN * 2);
  uint16_t* wpb  = (uint16_t*)carve((size_t)D_OUTD * D_OUTD * 2);
  uint16_t* qkvb = (uint16_t*)carve((size_t)MTOT * 3 * D_OUTD * 2);
  uint16_t* ctxb = (uint16_t*)carve((size_t)MTOT * D_OUTD * 2);
  (void)ws_size; (void)n_in; (void)in_sizes; (void)out_size;

  {
    int n1 = MTOT * D_IN;
    k_f32_to_bf16<<<(n1 + 255) / 256, 256, 0, stream>>>(x, xb, n1);
    int n2 = 3 * D_OUTD * D_IN;
    k_f32_to_bf16<<<(n2 + 255) / 256, 256, 0, stream>>>(w_qkv, wqb, n2);
    int n3 = D_OUTD * D_OUTD;
    k_f32_to_bf16<<<(n3 + 255) / 256, 256, 0, stream>>>(w_proj, wpb, n3);
  }

  // QKV: [4096,1024] x [3072,1024]^T -> bf16 [4096,3072]
  {
    dim3 g(MTOT / BM, (3 * D_OUTD) / BN);
    k_gemm<<<g, 256, 0, stream>>>(xb, wqb, qkvb, nullptr, nullptr, MTOT, 3 * D_OUTD, D_IN);
  }
  // causal flash attention -> ctx bf16 [4096,1024]
  {
    dim3 g(BATCH * NHEAD, SEQ / FBQ);
    k_flash<<<g, 128, 0, stream>>>(qkvb, ctxb);
  }
  // out: [4096,1024] x [1024,1024]^T + bias -> f32
  {
    dim3 g(MTOT / BM, D_OUTD / BN);
    k_gemm<<<g, 256, 0, stream>>>(ctxb, wpb, nullptr, out, b_proj, MTOT, D_OUTD, D_OUTD);
  }
}
